// GAT_83494164234982
// MI455X (gfx1250) — matrix operand order repaired and checked by reading, not compiled, not run on hardware
//
#include <hip/hip_runtime.h>
#include <hip/hip_bf16.h>

// ---------------------------------------------------------------------------
// GAT forward for MI455X (gfx1250, wave32, WMMA).
// B=32, N=1024, NFEAT=128, NHID=NOUT=64, NMLP=128, NCLASS=2.
// ---------------------------------------------------------------------------

typedef _Float16 v16h __attribute__((ext_vector_type(16)));
typedef _Float16 v8h  __attribute__((ext_vector_type(8)));
typedef _Float16 v4h  __attribute__((ext_vector_type(4)));
typedef float    v8f  __attribute__((ext_vector_type(8)));
typedef float    v4f  __attribute__((ext_vector_type(4)));
typedef int      v4i  __attribute__((ext_vector_type(4)));

// Address-space qualified vector pointers for the async-to-LDS builtin:
// signature is (v4i addrspace(1)* gsrc, v4i addrspace(3)* ldst, imm, imm).
typedef __attribute__((address_space(1))) v4i gv4i;
typedef __attribute__((address_space(3))) v4i lv4i;

#define B_    32
#define N_    1024
#define H_    64      // NHID == NOUT
#define NMLP_ 128
#define NCLS_ 2

#if __has_builtin(__builtin_amdgcn_global_load_async_to_lds_b128) && \
    __has_builtin(__builtin_amdgcn_s_wait_asynccnt)
#define HAVE_ASYNC_LDS 1
#else
#define HAVE_ASYNC_LDS 0
#endif

// Monotonic float -> unsigned key (total order), for u32 atomic max.
__device__ inline unsigned fkey(float f) {
  unsigned b = __float_as_uint(f);
  return (b & 0x80000000u) ? ~b : (b | 0x80000000u);
}
__device__ inline float funkey(unsigned k) {
  unsigned b = (k & 0x80000000u) ? (k & 0x7FFFFFFFu) : ~k;
  return __uint_as_float(b);
}

__device__ inline v16h cat8(v8h lo, v8h hi) {
  return __builtin_shufflevector(lo, hi, 0, 1, 2, 3, 4, 5, 6, 7,
                                 8, 9, 10, 11, 12, 13, 14, 15);
}

// ---------------------------------------------------------------------------
// Kernel 1: Wh = h @ W (f16 WMMA, fp32 accum); stores Wh TRANSPOSED
// (WhT[b][n][j], f16) so the aggregation's B fragments are contiguous.
// W is staged transposed into LDS (f16) once per block so B fragments are
// two ds_load_b128s. Fused f1 = Wh@a[:H], f2 = Wh@a[H:].
// One wave per 16-row tile; 4 waves per block share the W tile.
// ---------------------------------------------------------------------------
template <int KIN>
__global__ __launch_bounds__(128) void gat_transform(
    const float* __restrict__ h,     // [B,N,KIN]
    const float* __restrict__ W,     // [KIN,H]
    const float* __restrict__ avec,  // [2*H]
    _Float16* __restrict__ WhT,      // [B,H,N]  (transposed!)
    float* __restrict__ f1,          // [B,N]
    float* __restrict__ f2)          // [B,N]
{
  __shared__ __align__(16) _Float16 WT[H_][KIN];   // W transposed, f16

  const int tid  = threadIdx.x;
  const int lane = tid & 31;
  const int wave = tid >> 5;
  const int tile = blockIdx.x * 4 + wave;   // B*N/16 = 2048 tiles
  const int b    = tile >> 6;
  const int i0   = (tile & 63) << 4;
  const int hi   = lane >> 4;               // lane half (K-offset select)
  const int ln   = lane & 15;               // A row / B column within tile

  // Stage W^T (f16) cooperatively: coalesced global reads.
  for (int q = tid; q < KIN * H_; q += 128) {
    int k = q >> 6, n = q & 63;
    WT[n][k] = (_Float16)W[q];
  }
  __syncthreads();

  const float* hrow = h + ((size_t)(b * N_) + i0 + ln) * KIN;

  v8f c0 = {}, c1 = {}, c2 = {}, c3 = {};
#pragma unroll
  for (int kk = 0; kk < KIN; kk += 32) {
    // A fragment: 16-bit A 16x32 layout; both K runs are contiguous fp32.
    const v4f* r0 = (const v4f*)(hrow + kk + (hi ? 8 : 0));
    const v4f* r1 = (const v4f*)(hrow + kk + 16 + (hi ? 8 : 0));
    v4f a0f = r0[0], a1f = r0[1], a2f = r1[0], a3f = r1[1];
    v16h a;
#pragma unroll
    for (int e = 0; e < 4; ++e) {
      a[e]      = (_Float16)a0f[e];
      a[4 + e]  = (_Float16)a1f[e];
      a[8 + e]  = (_Float16)a2f[e];
      a[12 + e] = (_Float16)a3f[e];
    }
    // B fragments from LDS W^T: two 8-half K runs per column (same order as A).
    v16h bf[4];
#pragma unroll
    for (int t = 0; t < 4; ++t) {
      const _Float16* wr = &WT[t * 16 + ln][kk + (hi ? 8 : 0)];
      bf[t] = cat8(*(const v8h*)(wr), *(const v8h*)(wr + 16));
    }
    c0 = __builtin_amdgcn_wmma_f32_16x16x32_f16(false, a, false, bf[0], (short)0, c0, false, false);
    c1 = __builtin_amdgcn_wmma_f32_16x16x32_f16(false, a, false, bf[1], (short)0, c1, false, false);
    c2 = __builtin_amdgcn_wmma_f32_16x16x32_f16(false, a, false, bf[2], (short)0, c2, false, false);
    c3 = __builtin_amdgcn_wmma_f32_16x16x32_f16(false, a, false, bf[3], (short)0, c3, false, false);
  }

  // Epilogue: each lane owns 8 consecutive rows (M = hi*8 + r) of one output
  // column per n-tile -> contiguous along WhT's j axis: single b128 store.
  const v8f cacc[4] = {c0, c1, c2, c3};
  float pf1[8], pf2[8];
#pragma unroll
  for (int r = 0; r < 8; ++r) { pf1[r] = 0.f; pf2[r] = 0.f; }

#pragma unroll
  for (int t = 0; t < 4; ++t) {
    int n = t * 16 + ln;
    float a1 = avec[n];
    float a2 = avec[H_ + n];
    v8h st;
#pragma unroll
    for (int r = 0; r < 8; ++r) {
      float v = cacc[t][r];
      st[r] = (_Float16)v;
      pf1[r] += v * a1;
      pf2[r] += v * a2;
    }
    *(v8h*)(WhT + ((size_t)(b * H_) + n) * N_ + i0 + (hi ? 8 : 0)) = st;
  }
  // Reduce across the 16 lanes of each half (xor < 16 stays within half).
#pragma unroll
  for (int ofs = 1; ofs < 16; ofs <<= 1) {
#pragma unroll
    for (int r = 0; r < 8; ++r) {
      pf1[r] += __shfl_xor(pf1[r], ofs, 32);
      pf2[r] += __shfl_xor(pf2[r], ofs, 32);
    }
  }
  if (ln == 0) {
#pragma unroll
    for (int r = 0; r < 8; ++r) {
      int row = i0 + r + (hi ? 8 : 0);
      f1[b * N_ + row] = pf1[r];
      f2[b * N_ + row] = pf2[r];
    }
  }
}

// ---------------------------------------------------------------------------
// Kernel 2: masked-softmax attention + aggregation for one 16-row tile.
// adj tile (contiguous 64 KB) streamed once with v4f loads; probabilities
// live in LDS as f16; row max fused into the scoring pass (register-local).
// Aggregation attn @ Wh with 8 waves splitting K=1024, WMMA f32<=f16;
// B fragments are wide contiguous b128 loads of WhT (L2-resident).
// ---------------------------------------------------------------------------
__global__ __launch_bounds__(256) void gat_attend(
    const float* __restrict__ adj,      // [B,N,N]
    const float* __restrict__ f1,       // [B,N]
    const float* __restrict__ f2,       // [B,N]
    const _Float16* __restrict__ WhT,   // [B,H,N]
    float* __restrict__ hout)           // [B,N,H]  (ELU applied)
{
  __shared__ __align__(16) _Float16 p_lds[16][N_];  // 32 KB: scores -> probs
  __shared__ __align__(16) float f2s[N_];           // 4 KB
  __shared__ float    f1s[16];
  __shared__ unsigned rowmax_u[16];
  __shared__ float    rowsum[16];
  __shared__ float    rowinv[16];
  __shared__ __align__(16) float oacc[16][H_];      // 4 KB partial C reduce

  const int tid  = threadIdx.x;
  const int tile = blockIdx.x;
  const int b    = tile >> 6;
  const int i0   = (tile & 63) << 4;
  const float* adjt = adj + ((size_t)(b * N_) + i0) * N_;  // contiguous 16*N
  const int row = tid >> 4;       // row-owned mapping for score phases
  const int seg = tid & 15;       // 64-wide column segment within the row

  // ---- phase 0: stage f1/f2, init stats, prefetch the adj tile ----
#if HAVE_ASYNC_LDS
  // Each of the 256 threads asynchronously copies 16 B of f2 into LDS.
  {
    gv4i* gsrc = (gv4i*)(f2 + (size_t)b * N_ + tid * 4);
    lv4i* ldst = (lv4i*)(&f2s[tid * 4]);
    __builtin_amdgcn_global_load_async_to_lds_b128(gsrc, ldst, 0, 0);
  }
#endif
  if (tid < 16) {
    f1s[tid]      = f1[b * N_ + i0 + tid];
    rowmax_u[tid] = fkey(-3.0e38f);
    rowsum[tid]   = 0.f;
  }
  for (int q = tid; q < 16 * H_; q += 256) (&oacc[0][0])[q] = 0.f;
  for (int c = tid; c < 1024; c += 256)
    __builtin_prefetch(adjt + (size_t)c * 16, 0, 0);  // global_prefetch_b8
#if HAVE_ASYNC_LDS
  __builtin_amdgcn_s_wait_asynccnt(0);
#else
  for (int j = tid; j < N_; j += 256) f2s[j] = f2[b * N_ + j];
#endif
  __syncthreads();

  // ---- phase 1: masked leaky-relu scores -> LDS f16; fused row max ----
  {
    const float* arow = adjt + (size_t)row * N_ + seg * 64;
    const float* f2p  = &f2s[seg * 64];
    const float  f1r  = f1s[row];
    float lmax = -3.0e38f;
#pragma unroll 4
    for (int q = 0; q < 64; q += 4) {
      v4f av = *(const v4f*)(arow + q);
      v4f fv = *(const v4f*)(f2p + q);
      v4h st;
#pragma unroll
      for (int e = 0; e < 4; ++e) {
        float ev = f1r + fv[e];
        ev = ev > 0.f ? ev : 0.2f * ev;          // leaky relu, alpha = 0.2
        ev = av[e] > 0.f ? ev : -3.0e4f;         // f16-safe "-inf"
        lmax = fmaxf(lmax, ev);
        st[e] = (_Float16)ev;
      }
      *(v4h*)(&p_lds[row][seg * 64 + q]) = st;
    }
    atomicMax(&rowmax_u[row], fkey(lmax));       // ds_max_u32
  }
  __syncthreads();

  // ---- phase 1c: exp & row sum (unnormalized probs back to LDS f16) ----
  {
    float mx = funkey(rowmax_u[row]);
    float s = 0.f;
    _Float16* pp = &p_lds[row][seg * 64];
#pragma unroll 2
    for (int q = 0; q < 64; q += 8) {
      v8h pv = *(const v8h*)(pp + q);
      v8h ov;
#pragma unroll
      for (int e = 0; e < 8; ++e) {
        float p = __expf((float)pv[e] - mx);
        s += p;
        ov[e] = (_Float16)p;
      }
      *(v8h*)(pp + q) = ov;
    }
    atomicAdd(&rowsum[row], s);                  // ds_add_f32
  }
  __syncthreads();
  if (tid < 16) rowinv[tid] = 1.0f / rowsum[tid];
  __syncthreads();

  // ---- phase 2: attn @ Wh via WMMA; 8 waves split K = 1024 ----
  const int lane = tid & 31, wave = tid >> 5;
  const int hi = lane >> 4, ln = lane & 15;
  const _Float16* whb = WhT + (size_t)b * H_ * N_;

  v8f c0 = {}, c1 = {}, c2 = {}, c3 = {};
#pragma unroll
  for (int cc = 0; cc < 4; ++cc) {
    int kk = (wave + cc * 8) * 32;              // this wave's K chunk base
    // A fragment: two aligned contiguous 8-half runs (A 16x32 layout).
    const _Float16* prow = &p_lds[ln][0];
    v16h a = cat8(*(const v8h*)(prow + kk + (hi ? 8 : 0)),
                  *(const v8h*)(prow + kk + 16 + (hi ? 8 : 0)));
    // B fragments: two 8-half runs of WhT (column n), same K order as A.
    v16h bf[4];
#pragma unroll
    for (int t = 0; t < 4; ++t) {
      const _Float16* wr =
          whb + ((size_t)(t * 16 + ln)) * N_ + kk + (hi ? 8 : 0);
      bf[t] = cat8(*(const v8h*)(wr), *(const v8h*)(wr + 16));
    }
    c0 = __builtin_amdgcn_wmma_f32_16x16x32_f16(false, a, false, bf[0], (short)0, c0, false, false);
    c1 = __builtin_amdgcn_wmma_f32_16x16x32_f16(false, a, false, bf[1], (short)0, c1, false, false);
    c2 = __builtin_amdgcn_wmma_f32_16x16x32_f16(false, a, false, bf[2], (short)0, c2, false, false);
    c3 = __builtin_amdgcn_wmma_f32_16x16x32_f16(false, a, false, bf[3], (short)0, c3, false, false);
  }
  // Cross-wave reduction of partial accumulators (ds_add_f32).
#pragma unroll
  for (int r = 0; r < 8; ++r) {
    int m = r + (hi ? 8 : 0);
    atomicAdd(&oacc[m][ 0 + ln], c0[r]);
    atomicAdd(&oacc[m][16 + ln], c1[r]);
    atomicAdd(&oacc[m][32 + ln], c2[r]);
    atomicAdd(&oacc[m][48 + ln], c3[r]);
  }
  __syncthreads();

  // ---- phase 3: normalize (softmax denom), ELU, vectorized store ----
  {
    int orow = tid >> 4;          // 16 rows
    int n4   = (tid & 15) * 4;    // 4 consecutive features per thread
    float inv = rowinv[orow];
    v4f v = *(const v4f*)(&oacc[orow][n4]);
#pragma unroll
    for (int e = 0; e < 4; ++e) {
      float x = v[e] * inv;
      v[e] = x > 0.f ? x : (__expf(x) - 1.0f);   // ELU
    }
    *(v4f*)(hout + ((size_t)(b * N_) + i0 + orow) * H_ + n4) = v;
  }
}

// ---------------------------------------------------------------------------
// Kernel 3: masked mean readout  g[b,n] = sum_i h[b,i,n]*m[b,i] / max(sum m,1)
// ---------------------------------------------------------------------------
__global__ __launch_bounds__(256) void gat_readout(
    const float* __restrict__ h2,     // [B,N,H]
    const float* __restrict__ xmask,  // [B,N]
    float* __restrict__ g)            // [B,H]
{
  int t = blockIdx.x * 256 + threadIdx.x;  // 2048 = B*H threads
  int b = t >> 6, n = t & 63;
  const float* hb = h2 + ((size_t)b * N_) * H_ + n;
  const float* mb = xmask + b * N_;
  float s = 0.f, ms = 0.f;
  for (int i = 0; i < N_; ++i) {
    float m = mb[i];
    s  += hb[(size_t)i * H_] * m;
    ms += m;
  }
  g[b * H_ + n] = s / fmaxf(ms, 1.0f);
}

// ---------------------------------------------------------------------------
// Kernel 4: tiny MLP  out = relu(g@W1 + b1) @ W2 + b2
// ---------------------------------------------------------------------------
__global__ __launch_bounds__(256) void gat_mlp(
    const float* __restrict__ g,
    const float* __restrict__ W1, const float* __restrict__ b1,
    const float* __restrict__ W2, const float* __restrict__ b2,
    float* __restrict__ out)
{
  __shared__ float gs[B_][H_];
  __shared__ float zs[B_][NMLP_];
  int tid = threadIdx.x;
  for (int q = tid; q < B_ * H_; q += 256) (&gs[0][0])[q] = g[q];
  __syncthreads();
  for (int q = tid; q < B_ * NMLP_; q += 256) {
    int b = q >> 7, m = q & 127;
    float acc = b1[m];
    for (int k = 0; k < H_; ++k) acc += gs[b][k] * W1[k * NMLP_ + m];
    zs[b][m] = fmaxf(acc, 0.f);
  }
  __syncthreads();
  if (tid < B_ * NCLS_) {
    int b = tid >> 1, c = tid & 1;
    float acc = b2[c];
    for (int m = 0; m < NMLP_; ++m) acc += zs[b][m] * W2[m * NCLS_ + c];
    out[b * NCLS_ + c] = acc;
  }
}

// ---------------------------------------------------------------------------
extern "C" void kernel_launch(void* const* d_in, const int* in_sizes, int n_in,
                              void* d_out, int out_size, void* d_ws, size_t ws_size,
                              hipStream_t stream) {
  const float* x     = (const float*)d_in[0];
  const float* xmask = (const float*)d_in[1];
  const float* adj   = (const float*)d_in[2];
  const float* W0    = (const float*)d_in[3];
  const float* a0    = (const float*)d_in[4];
  const float* Wout  = (const float*)d_in[5];
  const float* aout  = (const float*)d_in[6];
  const float* W1    = (const float*)d_in[7];
  const float* b1    = (const float*)d_in[8];
  const float* W2    = (const float*)d_in[9];
  const float* b2    = (const float*)d_in[10];
  float* out = (float*)d_out;

  char* ws = (char*)d_ws;
  size_t off = 0;
  auto alloc = [&](size_t bytes) -> void* {
    void* p = ws + off;
    off = (off + bytes + 255) & ~(size_t)255;
    return p;
  };
  const size_t BN = (size_t)B_ * N_;
  _Float16* WhT0 = (_Float16*)alloc(BN * H_ * sizeof(_Float16));
  float*    f1_0 = (float*)   alloc(BN * sizeof(float));
  float*    f2_0 = (float*)   alloc(BN * sizeof(float));
  float*    h1   = (float*)   alloc(BN * H_ * sizeof(float));
  _Float16* WhT1 = (_Float16*)alloc(BN * H_ * sizeof(_Float16));
  float*    f1_1 = (float*)   alloc(BN * sizeof(float));
  float*    f2_1 = (float*)   alloc(BN * sizeof(float));
  float*    h2   = (float*)   alloc(BN * H_ * sizeof(float));
  float*    g    = (float*)   alloc((size_t)B_ * H_ * sizeof(float));

  const int tiles = B_ * (N_ / 16);  // 2048

  // Layer 1
  gat_transform<128><<<tiles / 4, 128, 0, stream>>>(x, W0, a0, WhT0, f1_0, f2_0);
  gat_attend<<<tiles, 256, 0, stream>>>(adj, f1_0, f2_0, WhT0, h1);
  // Layer 2
  gat_transform<64><<<tiles / 4, 128, 0, stream>>>(h1, Wout, aout, WhT1, f1_1, f2_1);
  gat_attend<<<tiles, 256, 0, stream>>>(adj, f1_1, f2_1, WhT1, h2);
  // Readout + MLP
  gat_readout<<<(B_ * H_) / 256, 256, 0, stream>>>(h2, xmask, g);
  gat_mlp<<<1, 256, 0, stream>>>(g, W1, b1, W2, b2, out);
}
